// ViewLearner_47768626266133
// MI455X (gfx1250) — matrix operand order repaired and checked by reading, not compiled, not run on hardware
//
#include <hip/hip_runtime.h>

typedef __attribute__((ext_vector_type(16))) _Float16 v16h;
typedef __attribute__((ext_vector_type(8)))  float    v8f;

#define D_FEAT 128
#define TWO_D  256
#define H_DIM  64

// One wave (32 lanes) computes 32 edge logits (two 16-row WMMA tiles):
//   h = relu([emb_src|emb_dst] @ W1 + b1);  logit = h @ W2 + b2
// GEMM1 runs on v_wmma_f32_16x16x32_f16 (f16 in, f32 accum). Each LDS-resident
// W1 B-fragment feeds TWO wmma ops (one per M-tile) to amortize ds loads.
__global__ __launch_bounds__(256) void edge_mlp_wmma_kernel(
    const float* __restrict__ node_emb,   // [N, 128]
    const int*   __restrict__ edge_index, // [2, E] flat
    const float* __restrict__ W1,         // [256, 64] row-major
    const float* __restrict__ b1,         // [64]
    const float* __restrict__ W2,         // [64]
    const float* __restrict__ b2,         // [1]
    float*       __restrict__ out,        // [E]
    int n_edges)
{
    // W1 packed as 32 B-fragments (8 K-blocks x 4 N-tiles), each 32 lanes x 16 f16.
    __shared__ alignas(32) _Float16 sB[8 * 4 * 32 * 16];   // 32 KB

    const int tid = threadIdx.x;

    // ---- Pack W1 (f32 -> f16) into LDS in WMMA B-fragment layout ----
    // Dense 16-bit B (32x16): lanes 0-15 hold K=0..7,16..23, lanes 16-31 hold K=8..15,24..31,
    // half i <-> k = 8*group + i (i<8), 16 + 8*group + (i-8) (i>=8); lane's 16 halfs stored contiguously.
    for (int s = tid; s < 1024; s += 256) {          // 32 frags * 32 lanes
        const int frag = s >> 5;
        const int ln   = s & 31;
        const int kb   = (frag >> 2) << 5;           // K block base (0,32,..,224)
        const int nt   = frag & 3;                   // N tile (0..3)
        const int k0   = kb + ((ln >> 4) << 3);      // +8 for lane group 1
        const int n    = nt * 16 + (ln & 15);
        _Float16* dstp = &sB[s * 16];
        #pragma unroll
        for (int i = 0; i < 16; ++i)
            dstp[i] = (_Float16)W1[(k0 + i + ((i >> 3) << 3)) * H_DIM + n];
    }
    __syncthreads();

    const int wave   = tid >> 5;
    const int lane   = tid & 31;
    const int tile32 = blockIdx.x * 8 + wave;        // 32 edges per wave
    const int ebase  = tile32 * 32;
    if (ebase >= n_edges) return;                    // wave-uniform exit

    const int  row  = lane & 15;                     // M row this lane feeds for A
    const int  sel  = lane >> 4;                     // lane group (K-half select)
    const int  e0   = min(ebase + row,      n_edges - 1);
    const int  e1   = min(ebase + 16 + row, n_edges - 1);
    const float* rowS0 = node_emb + (long)edge_index[e0] * D_FEAT;
    const float* rowD0 = node_emb + (long)edge_index[n_edges + e0] * D_FEAT;
    const float* rowS1 = node_emb + (long)edge_index[e1] * D_FEAT;
    const float* rowD1 = node_emb + (long)edge_index[n_edges + e1] * D_FEAT;

    // Fold b1 into accumulator init: C/D lane holds N = lane%16, same N for all 8 rows.
    const float bv0 = b1[row +  0];
    const float bv1 = b1[row + 16];
    const float bv2 = b1[row + 32];
    const float bv3 = b1[row + 48];
    v8f acc00, acc01, acc02, acc03;                  // M-tile 0
    v8f acc10, acc11, acc12, acc13;                  // M-tile 1
    #pragma unroll
    for (int r = 0; r < 8; ++r) {
        acc00[r] = bv0; acc01[r] = bv1; acc02[r] = bv2; acc03[r] = bv3;
        acc10[r] = bv0; acc11[r] = bv1; acc12[r] = bv2; acc13[r] = bv3;
    }

    // ---- K loop: 8 fragments of 16x32 over the 256-long concat row ----
    #pragma unroll
    for (int kb = 0; kb < TWO_D; kb += 32) {
        const float* base0 = (kb < D_FEAT) ? rowS0 : rowD0; // chunks never cross the 128 split
        const float* base1 = (kb < D_FEAT) ? rowS1 : rowD1;
        const int o0 = (kb & (D_FEAT - 1)) + sel * 8;       // K = kb + sel*8 + [0..8)

        const float4 p0 = *(const float4*)(base0 + o0);
        const float4 p1 = *(const float4*)(base0 + o0 + 4);
        const float4 p2 = *(const float4*)(base0 + o0 + 16); // K = kb+16+sel*8 + [0..8)
        const float4 p3 = *(const float4*)(base0 + o0 + 20);
        const float4 q0 = *(const float4*)(base1 + o0);
        const float4 q1 = *(const float4*)(base1 + o0 + 4);
        const float4 q2 = *(const float4*)(base1 + o0 + 16);
        const float4 q3 = *(const float4*)(base1 + o0 + 20);

        v16h a0, a1;
        a0[0]  = (_Float16)p0.x; a0[1]  = (_Float16)p0.y; a0[2]  = (_Float16)p0.z; a0[3]  = (_Float16)p0.w;
        a0[4]  = (_Float16)p1.x; a0[5]  = (_Float16)p1.y; a0[6]  = (_Float16)p1.z; a0[7]  = (_Float16)p1.w;
        a0[8]  = (_Float16)p2.x; a0[9]  = (_Float16)p2.y; a0[10] = (_Float16)p2.z; a0[11] = (_Float16)p2.w;
        a0[12] = (_Float16)p3.x; a0[13] = (_Float16)p3.y; a0[14] = (_Float16)p3.z; a0[15] = (_Float16)p3.w;
        a1[0]  = (_Float16)q0.x; a1[1]  = (_Float16)q0.y; a1[2]  = (_Float16)q0.z; a1[3]  = (_Float16)q0.w;
        a1[4]  = (_Float16)q1.x; a1[5]  = (_Float16)q1.y; a1[6]  = (_Float16)q1.z; a1[7]  = (_Float16)q1.w;
        a1[8]  = (_Float16)q2.x; a1[9]  = (_Float16)q2.y; a1[10] = (_Float16)q2.z; a1[11] = (_Float16)q2.w;
        a1[12] = (_Float16)q3.x; a1[13] = (_Float16)q3.y; a1[14] = (_Float16)q3.z; a1[15] = (_Float16)q3.w;

        const int fb = (kb >> 5) << 2;                      // fragment index base
        {
            const v16h bf = *(const v16h*)&sB[(((fb + 0) << 5) + lane) * 16];
            acc00 = __builtin_amdgcn_wmma_f32_16x16x32_f16(false, a0, false, bf, (short)0, acc00, false, false);
            acc10 = __builtin_amdgcn_wmma_f32_16x16x32_f16(false, a1, false, bf, (short)0, acc10, false, false);
        }
        {
            const v16h bf = *(const v16h*)&sB[(((fb + 1) << 5) + lane) * 16];
            acc01 = __builtin_amdgcn_wmma_f32_16x16x32_f16(false, a0, false, bf, (short)0, acc01, false, false);
            acc11 = __builtin_amdgcn_wmma_f32_16x16x32_f16(false, a1, false, bf, (short)0, acc11, false, false);
        }
        {
            const v16h bf = *(const v16h*)&sB[(((fb + 2) << 5) + lane) * 16];
            acc02 = __builtin_amdgcn_wmma_f32_16x16x32_f16(false, a0, false, bf, (short)0, acc02, false, false);
            acc12 = __builtin_amdgcn_wmma_f32_16x16x32_f16(false, a1, false, bf, (short)0, acc12, false, false);
        }
        {
            const v16h bf = *(const v16h*)&sB[(((fb + 3) << 5) + lane) * 16];
            acc03 = __builtin_amdgcn_wmma_f32_16x16x32_f16(false, a0, false, bf, (short)0, acc03, false, false);
            acc13 = __builtin_amdgcn_wmma_f32_16x16x32_f16(false, a1, false, bf, (short)0, acc13, false, false);
        }
    }

    // ---- Layer 2: relu + dot with W2, reduce across the 16-lane N groups ----
    const float w20 = W2[row +  0];
    const float w21 = W2[row + 16];
    const float w22 = W2[row + 32];
    const float w23 = W2[row + 48];
    const float bias2 = b2[0];

    #pragma unroll
    for (int mt = 0; mt < 2; ++mt) {
        const v8f& c0 = mt ? acc10 : acc00;
        const v8f& c1 = mt ? acc11 : acc01;
        const v8f& c2 = mt ? acc12 : acc02;
        const v8f& c3 = mt ? acc13 : acc03;
        float pr[8];
        #pragma unroll
        for (int r = 0; r < 8; ++r) {
            float p = fmaxf(c0[r], 0.f) * w20
                    + fmaxf(c1[r], 0.f) * w21
                    + fmaxf(c2[r], 0.f) * w22
                    + fmaxf(c3[r], 0.f) * w23;
            // butterfly within each 16-lane group (D rows: group0 -> M=r, group1 -> M=r+8)
            p += __shfl_xor(p, 1, 32);
            p += __shfl_xor(p, 2, 32);
            p += __shfl_xor(p, 4, 32);
            p += __shfl_xor(p, 8, 32);
            pr[r] = p + bias2;
        }
        if (row == 0) {                               // lanes 0 and 16 write 8 logits each
            const int obase = ebase + mt * 16 + sel * 8;
            #pragma unroll
            for (int r = 0; r < 8; ++r)
                if (obase + r < n_edges) out[obase + r] = pr[r];
        }
    }
}

extern "C" void kernel_launch(void* const* d_in, const int* in_sizes, int n_in,
                              void* d_out, int out_size, void* d_ws, size_t ws_size,
                              hipStream_t stream) {
    const float* node_emb   = (const float*)d_in[0];
    const int*   edge_index = (const int*)  d_in[1];
    const float* W1         = (const float*)d_in[2];
    const float* b1         = (const float*)d_in[3];
    const float* W2         = (const float*)d_in[4];
    const float* b2         = (const float*)d_in[5];
    float*       out        = (float*)d_out;

    const int n_edges = out_size;                  // [E,1] logits
    const int tiles32 = (n_edges + 31) / 32;       // 32 edges per wave
    const int blocks  = (tiles32 + 7) / 8;         // 8 waves per 256-thread block
    edge_mlp_wmma_kernel<<<blocks, 256, 0, stream>>>(node_emb, edge_index, W1, b1, W2, b2, out, n_edges);
}
